// GAT_58231166599541
// MI455X (gfx1250) — compile-verified, matrix operand order checked by reading, not run on hardware
//
#include <hip/hip_runtime.h>
#include <hip/hip_bf16.h>
#include <math.h>

// GATv2 (3 layers) for MI455X / gfx1250, wave32.
// Dense transforms: v_wmma_f32_16x16x32_f16, weights pre-packed into B-fragment
// layout so the GEMM inner loop is pure b128 loads + WMMA.
// Graph softmax/aggregation: 3 edge passes with L2-resident gathers + atomics.

#define NEG_SLOPE 0.2f

typedef __attribute__((ext_vector_type(16))) _Float16 v16h;
typedef __attribute__((ext_vector_type(8)))  _Float16 v8h;
typedef __attribute__((ext_vector_type(8)))  float    v8f;

// ---- order-preserving float<->uint encoding for atomic max on f32 ----
__device__ __forceinline__ unsigned fenc(float f) {
  unsigned u = __float_as_uint(f);
  return (u & 0x80000000u) ? ~u : (u | 0x80000000u);
}
__device__ __forceinline__ float fdec(unsigned u) {
  return (u & 0x80000000u) ? __uint_as_float(u & 0x7fffffffu) : __uint_as_float(~u);
}
#define ENC_NEG_INF 0x007FFFFFu   // fenc(-inf)

// ---------------- elementwise helpers ----------------
__global__ void k_f32_to_f16(const float* __restrict__ src, _Float16* __restrict__ dst,
                             long long n) {
  long long i = (long long)blockIdx.x * blockDim.x + threadIdx.x;
  if (i < n) dst[i] = (_Float16)src[i];
}

__global__ void k_init_out(float* __restrict__ out, const float* __restrict__ bias,
                           long long n, int hc) {
  long long i = (long long)blockIdx.x * blockDim.x + threadIdx.x;
  if (i < n) out[i] = bias[(int)(i % hc)];
}

__global__ void k_init_md(unsigned* __restrict__ M, float* __restrict__ D, long long n) {
  long long i = (long long)blockIdx.x * blockDim.x + threadIdx.x;
  if (i < n) { M[i] = ENC_NEG_INF; D[i] = 0.0f; }
}

__global__ void k_elu_f16(const float* __restrict__ in, _Float16* __restrict__ out,
                          long long n) {
  long long i = (long long)blockIdx.x * blockDim.x + threadIdx.x;
  if (i >= n) return;
  float v = in[i];
  v = (v > 0.0f) ? v : expm1f(v);
  out[i] = (_Float16)v;
}

// ---------------- weight pre-pack: f32 W[din x dout] -> B-fragment layout ----------------
// Fragment (kt, nt) holds the 32x16 f16 block W[kt*32 .. +32)[nt*16 .. +16) in the
// CDNA5 ISA 7.12.2 16-bit B layout:
//   lanes 0-15 : K = 0..7 (vgpr0-3), 16..23 (vgpr4-7);  lanes 16-31: K = 8..15, 24..31
// Stored as WP[(kt*ntiles + nt)*32 + lane][16 halfs] so each lane's fragment is one
// contiguous 32-byte chunk (two global_load_b128 in the GEMM).
__global__ void k_pack_w(const float* __restrict__ W, _Float16* __restrict__ WP,
                         int din, int dout) {
  int ktiles = din >> 5, ntiles = dout >> 4;
  int total = ktiles * ntiles * 32;
  int idx = blockIdx.x * blockDim.x + threadIdx.x;
  if (idx >= total) return;
  int lane = idx & 31;
  int frag = idx >> 5;
  int nt = frag % ntiles;
  int kt = frag / ntiles;
  int half = lane >> 4, lr = lane & 15;
  _Float16* o = WP + ((size_t)frag * 32 + lane) * 16;
  const float* w0 = W + (size_t)(kt * 32 + half * 8) * dout + nt * 16 + lr;
#pragma unroll
  for (int j = 0; j < 8; ++j) {
    o[j]     = (_Float16)w0[(size_t)j * dout];
    o[8 + j] = (_Float16)w0[(size_t)(16 + j) * dout];
  }
}

// ---------------- WMMA GEMM: Y[M x dout] = Xh[M x din] @ W(packed) ----------------
// Block = 128 threads (4 waves). Each wave computes a 16x64 strip: one A fragment
// feeds 4 WMMAs per k-step. B fragments come from the packed layout (contiguous).
__global__ void k_gemm_f16(const _Float16* __restrict__ X, const _Float16* __restrict__ WP,
                           float* __restrict__ Y, int Mrows, int din, int dout) {
  int wave = threadIdx.x >> 5;
  int lane = threadIdx.x & 31;
  int half = lane >> 4;        // which K-halfgroup this lane holds
  int lr   = lane & 15;        // row (A) / col (B) within the 16-tile
  int tileM = blockIdx.x * 4 + wave;
  if (tileM * 16 >= Mrows) return;          // wave-uniform: EXEC stays all-ones
  int nt0 = blockIdx.y * 4;                 // first of 4 col tiles (64 cols)
  int ntiles = dout >> 4;

  int row = tileM * 16 + lr;
  if (row >= Mrows) row = Mrows - 1;        // clamp (N is an exact multiple of 16 here)

  v8f acc0 = {}, acc1 = {}, acc2 = {}, acc3 = {};
  const _Float16* xrow = X + (size_t)row * din;
  const v16h* WPv = (const v16h*)WP;
  for (int k0 = 0; k0 < din; k0 += 32) {
    // A fragment: two aligned 16B vector loads, register concat (no repacking)
    const v8h* xa = (const v8h*)(xrow + k0 + half * 8);
    v8h lo = xa[0];
    v8h hi = xa[2];
    v16h a = __builtin_shufflevector(lo, hi, 0, 1, 2, 3, 4, 5, 6, 7,
                                     8, 9, 10, 11, 12, 13, 14, 15);
    // B fragments: contiguous 32B per lane from the packed layout
    size_t fb = ((size_t)(k0 >> 5) * ntiles + nt0) * 32 + lane;
    v16h b0 = WPv[fb];
    v16h b1 = WPv[fb + 32];
    v16h b2 = WPv[fb + 64];
    v16h b3 = WPv[fb + 96];
    acc0 = __builtin_amdgcn_wmma_f32_16x16x32_f16(false, a, false, b0, (short)0, acc0, false, false);
    acc1 = __builtin_amdgcn_wmma_f32_16x16x32_f16(false, a, false, b1, (short)0, acc1, false, false);
    acc2 = __builtin_amdgcn_wmma_f32_16x16x32_f16(false, a, false, b2, (short)0, acc2, false, false);
    acc3 = __builtin_amdgcn_wmma_f32_16x16x32_f16(false, a, false, b3, (short)0, acc3, false, false);
  }
  // C/D layout: lanes 0-15 -> rows 0..7 (vgpr i == row i), lanes 16-31 -> rows 8..15
  int r0 = tileM * 16 + half * 8;
#pragma unroll
  for (int i = 0; i < 8; ++i) {
    int r = r0 + i;
    if (r < Mrows) {
      float* yp = Y + (size_t)r * dout + nt0 * 16 + lr;
      yp[0]  = acc0[i];
      yp[16] = acc1[i];
      yp[32] = acc2[i];
      yp[48] = acc3[i];
    }
  }
}

// ---------------- edge pass 1: scores + segment max ----------------
// One wave per edge; 32/H lanes per head; shfl-reduce the per-head dot product.
template <int H, int C>
__global__ void k_edge_score(const float* __restrict__ XL, const float* __restrict__ XR,
                             const float* __restrict__ att, const int* __restrict__ ei,
                             long long E0, long long ET,
                             float* __restrict__ SC, unsigned* __restrict__ M) {
  constexpr int HC  = H * C;
  constexpr int LPH = 32 / H;       // lanes per head
  constexpr int CPT = C / LPH;      // channels per thread
  long long e = (long long)blockIdx.x * (blockDim.x >> 5) + (threadIdx.x >> 5);
  if (e >= ET) return;
  int lane = threadIdx.x & 31;
  int s, d;
  if (e < E0) { s = ei[e]; d = ei[E0 + e]; }
  else        { s = d = (int)(e - E0); }    // self loop

  int h  = lane / LPH;
  int c0 = (lane % LPH) * CPT;
  const float* xl = XL + (size_t)s * HC + h * C + c0;
  const float* xr = XR + (size_t)d * HC + h * C + c0;
  const float* ah = att + h * C + c0;
  float p = 0.0f;
#pragma unroll
  for (int j = 0; j < CPT; ++j) {
    float v = xl[j] + xr[j];
    v = (v > 0.0f) ? v : NEG_SLOPE * v;     // LeakyReLU
    p += ah[j] * v;
  }
#pragma unroll
  for (int o = LPH >> 1; o > 0; o >>= 1) p += __shfl_xor(p, o, 32);
  if ((lane % LPH) == 0) {
    SC[(size_t)e * H + h] = p;
    atomicMax(&M[(size_t)d * H + h], fenc(p));
  }
}

// ---------------- edge pass 2: exp(score - max) + segment sum ----------------
template <int H>
__global__ void k_edge_exp(float* __restrict__ SC, const unsigned* __restrict__ M,
                           float* __restrict__ DEN, const int* __restrict__ ei,
                           long long E0, long long ET) {
  long long i = (long long)blockIdx.x * blockDim.x + threadIdx.x;
  if (i >= ET * (long long)H) return;
  long long e = i / H;
  int h = (int)(i % H);
  int d = (e < E0) ? ei[E0 + e] : (int)(e - E0);
  float m = fdec(M[(size_t)d * H + h]);
  if (!(m > -3.0e38f && m < 3.0e38f)) m = 0.0f;   // mirror where(isfinite(m), m, 0)
  float ex = __expf(SC[i] - m);
  SC[i] = ex;
  atomicAdd(&DEN[(size_t)d * H + h], ex);
}

// ---------------- edge pass 3: alpha-weighted scatter-add of xl[src] ----------------
template <int H, int C>
__global__ void k_edge_aggr(const float* __restrict__ XL, const float* __restrict__ SC,
                            const float* __restrict__ DEN, float* __restrict__ OUT,
                            const int* __restrict__ ei, long long E0, long long ET) {
  constexpr int HC  = H * C;
  constexpr int CPT = HC / 32;      // channels per thread (4 or 2), stays within one head
  long long e = (long long)blockIdx.x * (blockDim.x >> 5) + (threadIdx.x >> 5);
  if (e >= ET) return;
  int lane = threadIdx.x & 31;
  int s, d;
  if (e < E0) { s = ei[e]; d = ei[E0 + e]; }
  else        { s = d = (int)(e - E0); }
  int ch0 = lane * CPT;
  int h   = ch0 / C;
  float ex    = SC[(size_t)e * H + h];
  float dn    = DEN[(size_t)d * H + h];
  float alpha = ex / fmaxf(dn, 1e-16f);
  const float* xs = XL + (size_t)s * HC + ch0;
  float*       od = OUT + (size_t)d * HC + ch0;
#pragma unroll
  for (int j = 0; j < CPT; ++j) atomicAdd(&od[j], alpha * xs[j]);
}

// ---------------- final row softmax over 64 channels (one wave per node) ----------------
__global__ void k_softmax64(const float* __restrict__ in, float* __restrict__ out, int n) {
  int node = blockIdx.x * (blockDim.x >> 5) + (threadIdx.x >> 5);
  if (node >= n) return;
  int lane = threadIdx.x & 31;
  const float* p = in + (size_t)node * 64 + lane * 2;
  float a = p[0], b = p[1];
  float mx = fmaxf(a, b);
#pragma unroll
  for (int o = 16; o > 0; o >>= 1) mx = fmaxf(mx, __shfl_xor(mx, o, 32));
  float ea = __expf(a - mx), eb = __expf(b - mx);
  float sum = ea + eb;
#pragma unroll
  for (int o = 16; o > 0; o >>= 1) sum += __shfl_xor(sum, o, 32);
  float inv = 1.0f / sum;
  float* q = out + (size_t)node * 64 + lane * 2;
  q[0] = ea * inv;
  q[1] = eb * inv;
}

// ---------------- one GATv2 layer ----------------
template <int H, int C>
static void gat_layer(const _Float16* X16, int din,
                      const float* Wl, const float* Wr, const float* att, const float* b,
                      _Float16* WPa, _Float16* WPb,
                      float* XL, float* XR, float* OUT, unsigned* M, float* DEN, float* SC,
                      const int* ei, int N, long long E0, hipStream_t stream) {
  constexpr int HC = H * C;
  long long ET = E0 + N;

  // pack (and convert) weights into B-fragment layout
  int packThreads = (din >> 5) * (HC >> 4) * 32;
  int packBlocks  = (packThreads + 255) / 256;
  k_pack_w<<<packBlocks, 256, 0, stream>>>(Wl, WPa, din, HC);
  k_pack_w<<<packBlocks, 256, 0, stream>>>(Wr, WPb, din, HC);

  int rowTiles = (N + 15) / 16;
  dim3 gg((unsigned)((rowTiles + 3) / 4), (unsigned)(HC / 64));
  k_gemm_f16<<<gg, 128, 0, stream>>>(X16, WPa, XL, N, din, HC);
  k_gemm_f16<<<gg, 128, 0, stream>>>(X16, WPb, XR, N, din, HC);

  long long nHC = (long long)N * HC, nH = (long long)N * H;
  k_init_out<<<(int)((nHC + 255) / 256), 256, 0, stream>>>(OUT, b, nHC, HC);
  k_init_md<<<(int)((nH + 255) / 256), 256, 0, stream>>>(M, DEN, nH);

  int gE = (int)((ET + 7) / 8);   // 8 waves (edges) per 256-thread block
  k_edge_score<H, C><<<gE, 256, 0, stream>>>(XL, XR, att, ei, E0, ET, SC, M);
  k_edge_exp<H><<<(int)((ET * H + 255) / 256), 256, 0, stream>>>(SC, M, DEN, ei, E0, ET);
  k_edge_aggr<H, C><<<gE, 256, 0, stream>>>(XL, SC, DEN, OUT, ei, E0, ET);
}

extern "C" void kernel_launch(void* const* d_in, const int* in_sizes, int n_in,
                              void* d_out, int out_size, void* d_ws, size_t ws_size,
                              hipStream_t stream) {
  (void)n_in; (void)out_size; (void)ws_size;
  const float* x    = (const float*)d_in[0];
  const int*   ei   = (const int*)d_in[1];   // edge_index flattened [2, E0] (int)
  const float* Wl1  = (const float*)d_in[2];
  const float* Wr1  = (const float*)d_in[3];
  const float* att1 = (const float*)d_in[4];
  const float* b1   = (const float*)d_in[5];
  const float* Wl2  = (const float*)d_in[6];
  const float* Wr2  = (const float*)d_in[7];
  const float* att2 = (const float*)d_in[8];
  const float* b2   = (const float*)d_in[9];
  const float* Wl3  = (const float*)d_in[10];
  const float* Wr3  = (const float*)d_in[11];
  const float* att3 = (const float*)d_in[12];
  const float* b3   = (const float*)d_in[13];

  int       N  = in_sizes[0] / 256;
  long long E0 = (long long)in_sizes[1] / 2;
  long long ET = E0 + N;

  // workspace carve (256B aligned)
  uintptr_t p = (uintptr_t)d_ws;
  auto carve = [&](size_t bytes) -> void* {
    void* r = (void*)p;
    p += (bytes + 255) & ~(size_t)255;
    return r;
  };
  _Float16* X16 = (_Float16*)carve((size_t)N * 256 * sizeof(_Float16));
  _Float16* WPa = (_Float16*)carve((size_t)256 * 128 * sizeof(_Float16));
  _Float16* WPb = (_Float16*)carve((size_t)256 * 128 * sizeof(_Float16));
  float*    XL  = (float*)carve((size_t)N * 128 * sizeof(float));
  float*    XR  = (float*)carve((size_t)N * 128 * sizeof(float));
  float*    OUT = (float*)carve((size_t)N * 128 * sizeof(float));
  unsigned* M   = (unsigned*)carve((size_t)N * 8 * sizeof(unsigned));
  float*    DEN = (float*)carve((size_t)N * 8 * sizeof(float));
  float*    SC  = (float*)carve((size_t)ET * 8 * sizeof(float));

  // layer 1: 256 -> 8x16
  long long nx = (long long)N * 256;
  k_f32_to_f16<<<(int)((nx + 255) / 256), 256, 0, stream>>>(x, X16, nx);
  gat_layer<8, 16>(X16, 256, Wl1, Wr1, att1, b1, WPa, WPb,
                   XL, XR, OUT, M, DEN, SC, ei, N, E0, stream);
  long long nh = (long long)N * 128;
  k_elu_f16<<<(int)((nh + 255) / 256), 256, 0, stream>>>(OUT, X16, nh);

  // layer 2: 128 -> 8x16
  gat_layer<8, 16>(X16, 128, Wl2, Wr2, att2, b2, WPa, WPb,
                   XL, XR, OUT, M, DEN, SC, ei, N, E0, stream);
  k_elu_f16<<<(int)((nh + 255) / 256), 256, 0, stream>>>(OUT, X16, nh);

  // layer 3: 128 -> 1x64
  gat_layer<1, 64>(X16, 128, Wl3, Wr3, att3, b3, WPa, WPb,
                   XL, XR, OUT, M, DEN, SC, ei, N, E0, stream);

  // row softmax over 64 channels -> d_out
  k_softmax64<<<(N + 7) / 8, 256, 0, stream>>>(OUT, (float*)d_out, N);
}
